// GraphEncoder_14602888807204
// MI455X (gfx1250) — matrix operand order repaired and checked by reading, not compiled, not run on hardware
//
#include <hip/hip_runtime.h>
#include <math.h>

#define BATCH 2
#define NNODE 20000
#define NEDGE 160000
#define XDIM  128
#define H1DIM 1024
#define H2DIM 512
#define HDDIM 256
#define ZDIM  128
#define BN    (BATCH * NNODE)   // 40000 rows
#define MTILES (BN / 16)        // 2500
#define NCHUNK 32
#define NSTEP  625              // 625*32 == 20000

typedef __attribute__((ext_vector_type(16))) __bf16 bf16x16;
typedef __attribute__((ext_vector_type(8)))  float  f32x8;

union Frag { uint4 q[2]; bf16x16 v; };

__device__ __forceinline__ unsigned short f2bf(float f) {
  union { float f; unsigned u; } v; v.f = f;
  unsigned r = v.u + 0x7FFFu + ((v.u >> 16) & 1u);
  return (unsigned short)(r >> 16);
}
__device__ __forceinline__ float bf2f(unsigned h) {
  union { unsigned u; float f; } v; v.u = (h & 0xFFFFu) << 16; return v.f;
}

// ---------------- degree / edge norm ----------------
__global__ void k_deg_init(float* deg) {
  int i = blockIdx.x * blockDim.x + threadIdx.x;
  if (i < NNODE) deg[i] = 1.0f;                      // self-loop weight
}
__global__ void k_deg_acc(const int* __restrict__ ecol, const float* __restrict__ w,
                          float* __restrict__ deg) {
  int e = blockIdx.x * blockDim.x + threadIdx.x;
  if (e < NEDGE) atomicAdd(&deg[ecol[e]], w[e]);
}
__global__ void k_dinv(const float* __restrict__ deg, float* __restrict__ dinv) {
  int i = blockIdx.x * blockDim.x + threadIdx.x;
  if (i < NNODE) dinv[i] = rsqrtf(deg[i]);
}
__global__ void k_enorm(const int* __restrict__ erow, const int* __restrict__ ecol,
                        const float* __restrict__ w, const float* __restrict__ dinv,
                        float* __restrict__ en) {
  int e = blockIdx.x * blockDim.x + threadIdx.x;
  if (e < NEDGE) en[e] = dinv[erow[e]] * w[e] * dinv[ecol[e]];
}

// ---------------- f32 -> bf16 convert (x) ----------------
__global__ void k_cvt_x(const float* __restrict__ x, unsigned short* __restrict__ o) {
  size_t i = ((size_t)blockIdx.x * blockDim.x + threadIdx.x) * 4;
  if (i >= (size_t)BN * XDIM) return;
  float4 v = *(const float4*)(x + i);
  uint2 p;
  p.x = (unsigned)f2bf(v.x) | ((unsigned)f2bf(v.y) << 16);
  p.y = (unsigned)f2bf(v.z) | ((unsigned)f2bf(v.w) << 16);
  *(uint2*)(o + i) = p;
}

// ---------------- pack weight into WMMA B-fragment layout ----------------
// Bp[((no*Ktiles+ko)*32 + lane)*16 + i] = W[(ko*32 + 16*(lane>>4) + i) * Fout + no*16 + (lane&15)]
__global__ void k_pack_w(const float* __restrict__ W, unsigned short* __restrict__ Wp,
                         int K, int Fout) {
  int tid = blockIdx.x * blockDim.x + threadIdx.x;
  int Ktiles = K >> 5, Ntiles = Fout >> 4;
  if (tid >= Ntiles * Ktiles * 32) return;
  int lane = tid & 31;
  int ko = (tid >> 5) % Ktiles;
  int no = tid / (32 * Ktiles);
  int n = (no << 4) + (lane & 15);
  int kbase = (ko << 5) + ((lane >> 4) << 4);
  unsigned short* dst = Wp + (size_t)tid * 16;
  for (int i = 0; i < 16; ++i)
    dst[i] = f2bf(W[(size_t)(kbase + i) * Fout + n]);
}

// ---------------- WMMA GEMM: C[M,Fout](bf16) = A[M,K](bf16) x W(packed bf16) ----------------
__global__ void k_gemm_bf16(const unsigned short* __restrict__ A,
                            const unsigned short* __restrict__ Bp,
                            unsigned short* __restrict__ C,
                            int K, int Fout) {
  const int lane = threadIdx.x & 31;
  const int wave = threadIdx.x >> 5;
  const int mt = blockIdx.x * 8 + wave;
  if (mt >= MTILES) return;                        // wave-uniform: EXEC stays all-ones
  const int ng = blockIdx.y;                       // group of 4 N-tiles
  const int Ktiles = K >> 5;
  const int half = lane >> 4;
  const int mrow = (mt << 4) + (lane & 15);
  const unsigned short* arow = A + (size_t)mrow * K + (half << 4);
  const size_t bstride = (size_t)Ktiles * 32 * 16;
  const unsigned short* bbase =
      Bp + ((size_t)(ng * 4) * Ktiles * 32 + lane) * 16;

  f32x8 acc0 = {}, acc1 = {}, acc2 = {}, acc3 = {};
  for (int ko = 0; ko < Ktiles; ++ko) {
    Frag a;
    const unsigned short* ap = arow + (ko << 5);
    a.q[0] = *(const uint4*)(ap);                  // K = 16*half + 0..7
    a.q[1] = *(const uint4*)(ap + 8);              // K = 16*half + 8..15
    __builtin_prefetch(ap + 32, 0, 1);             // next K tile of A

    const unsigned short* bp = bbase + (size_t)ko * 32 * 16;
    Frag b0, b1, b2, b3;
    b0.q[0] = *(const uint4*)(bp);               b0.q[1] = *(const uint4*)(bp + 8);
    b1.q[0] = *(const uint4*)(bp + bstride);     b1.q[1] = *(const uint4*)(bp + bstride + 8);
    b2.q[0] = *(const uint4*)(bp + 2 * bstride); b2.q[1] = *(const uint4*)(bp + 2 * bstride + 8);
    b3.q[0] = *(const uint4*)(bp + 3 * bstride); b3.q[1] = *(const uint4*)(bp + 3 * bstride + 8);

    acc0 = __builtin_amdgcn_wmma_f32_16x16x32_bf16(false, a.v, false, b0.v, (short)0, acc0, false, false);
    acc1 = __builtin_amdgcn_wmma_f32_16x16x32_bf16(false, a.v, false, b1.v, (short)0, acc1, false, false);
    acc2 = __builtin_amdgcn_wmma_f32_16x16x32_bf16(false, a.v, false, b2.v, (short)0, acc2, false, false);
    acc3 = __builtin_amdgcn_wmma_f32_16x16x32_bf16(false, a.v, false, b3.v, (short)0, acc3, false, false);
  }
  // C/D layout: lane -> col = lane&15, VGPR r -> row = r + 8*(lane>>4)
  const int colb = (ng * 4) << 4;
  const int col = colb + (lane & 15);
  const int rbase = (mt << 4) + (half << 3);
  for (int r = 0; r < 8; ++r) {
    size_t ro = (size_t)(rbase + r) * Fout;
    C[ro + col]      = f2bf(acc0[r]);
    C[ro + col + 16] = f2bf(acc1[r]);
    C[ro + col + 32] = f2bf(acc2[r]);
    C[ro + col + 48] = f2bf(acc3[r]);
  }
}

// ---------------- aggregation ----------------
__global__ void k_agg_init(const unsigned short* __restrict__ h, const float* __restrict__ dinv,
                           float* __restrict__ agg, int F) {
  int chunks = F >> 2;
  size_t idx = (size_t)blockIdx.x * blockDim.x + threadIdx.x;
  if (idx >= (size_t)BN * chunks) return;
  int c = (int)(idx % chunks);
  int m = (int)(idx / chunks);
  int n = m % NNODE;
  float dv = dinv[n], s = dv * dv;                 // self-loop coefficient
  uint2 u = *(const uint2*)(h + (size_t)m * F + (c << 2));
  float4 o;
  o.x = s * bf2f(u.x); o.y = s * bf2f(u.x >> 16);
  o.z = s * bf2f(u.y); o.w = s * bf2f(u.y >> 16);
  *(float4*)(agg + (size_t)m * F + (c << 2)) = o;
}

__global__ void k_agg_edges(const unsigned short* __restrict__ h, const int* __restrict__ erow,
                            const int* __restrict__ ecol, const float* __restrict__ en,
                            float* __restrict__ agg, int F) {
  int chunks = F >> 2;
  size_t idx = (size_t)blockIdx.x * blockDim.x + threadIdx.x;
  if (idx >= (size_t)BATCH * NEDGE * chunks) return;
  int c = (int)(idx % chunks);
  size_t t = idx / chunks;
  int e = (int)(t % NEDGE);
  int b = (int)(t / NEDGE);
  int r = erow[e], d = ecol[e];
  float w = en[e];
  uint2 u = *(const uint2*)(h + ((size_t)(b * NNODE + r)) * F + (c << 2));
  float* dp = agg + ((size_t)(b * NNODE + d)) * F + (c << 2);
  atomicAdd(dp + 0, w * bf2f(u.x));
  atomicAdd(dp + 1, w * bf2f(u.x >> 16));
  atomicAdd(dp + 2, w * bf2f(u.y));
  atomicAdd(dp + 3, w * bf2f(u.y >> 16));
}

// ---------------- l2norm over node axis ----------------
__global__ void k_sumsq(const float* __restrict__ agg, const float* __restrict__ bias,
                        float* __restrict__ partial, int F) {
  int tid = blockIdx.x * blockDim.x + threadIdx.x;
  if (tid >= BATCH * F * NCHUNK) return;
  int f = tid % F;
  int ck = (tid / F) & (NCHUNK - 1);
  int b = tid / (F * NCHUNK);
  const float* base = agg + ((size_t)b * NNODE + (size_t)ck * NSTEP) * F + f;
  float bi = bias[f], acc = 0.f;
  for (int n = 0; n < NSTEP; ++n) {
    float y = base[(size_t)n * F] + bi;
    acc += y * y;
  }
  partial[tid] = acc;
}
__global__ void k_colinv(const float* __restrict__ partial, float* __restrict__ cninv, int F) {
  int tid = blockIdx.x * blockDim.x + threadIdx.x;
  if (tid >= BATCH * F) return;
  int f = tid % F, b = tid / F;
  const float* p = partial + (size_t)b * NCHUNK * F + f;
  float s = 0.f;
  for (int ck = 0; ck < NCHUNK; ++ck) s += p[(size_t)ck * F];
  cninv[tid] = 1.0f / fmaxf(sqrtf(s), 1e-12f);
}
__global__ void k_normrelu(const float* __restrict__ agg, const float* __restrict__ bias,
                           const float* __restrict__ cninv, unsigned short* __restrict__ out, int F) {
  int chunks = F >> 2;
  size_t idx = (size_t)blockIdx.x * blockDim.x + threadIdx.x;
  if (idx >= (size_t)BN * chunks) return;
  int c = (int)(idx % chunks);
  int m = (int)(idx / chunks);
  int b = m / NNODE;
  int f = c << 2;
  float4 v = *(const float4*)(agg + (size_t)m * F + f);
  float4 bi = *(const float4*)(bias + f);
  float4 iv = *(const float4*)(cninv + b * F + f);
  float y0 = fmaxf((v.x + bi.x) * iv.x, 0.f);
  float y1 = fmaxf((v.y + bi.y) * iv.y, 0.f);
  float y2 = fmaxf((v.z + bi.z) * iv.z, 0.f);
  float y3 = fmaxf((v.w + bi.w) * iv.w, 0.f);
  uint2 p;
  p.x = (unsigned)f2bf(y0) | ((unsigned)f2bf(y1) << 16);
  p.y = (unsigned)f2bf(y2) | ((unsigned)f2bf(y3) << 16);
  *(uint2*)(out + (size_t)m * F + f) = p;
}

// ---------------- pooling ----------------
__global__ void k_pool_partial(const unsigned short* __restrict__ act, float* __restrict__ partial) {
  int tid = blockIdx.x * blockDim.x + threadIdx.x;
  if (tid >= BATCH * HDDIM * NCHUNK) return;
  int f = tid % HDDIM;
  int ck = (tid / HDDIM) & (NCHUNK - 1);
  int b = tid / (HDDIM * NCHUNK);
  const unsigned short* base = act + ((size_t)b * NNODE + (size_t)ck * NSTEP) * HDDIM + f;
  float acc = 0.f;
  for (int n = 0; n < NSTEP; ++n) acc += bf2f(base[(size_t)n * HDDIM]);
  partial[tid] = acc;
}
__global__ void k_pool_final(const float* __restrict__ partial, float* __restrict__ g) {
  int tid = blockIdx.x * blockDim.x + threadIdx.x;
  if (tid >= BATCH * HDDIM) return;
  int f = tid % HDDIM, b = tid / HDDIM;
  const float* p = partial + (size_t)b * NCHUNK * HDDIM + f;
  float s = 0.f;
  for (int ck = 0; ck < NCHUNK; ++ck) s += p[(size_t)ck * HDDIM];
  g[tid] = s;
}

// ---------------- heads: GEMV + BatchNorm(B=2) + relu (+ sigmoid) ----------------
__global__ void k_head(const float* __restrict__ g, const float* __restrict__ yt,
                       const float* __restrict__ Wm, const float* __restrict__ bm,
                       const float* __restrict__ gm, const float* __restrict__ bem,
                       const float* __restrict__ Wv, const float* __restrict__ bv,
                       const float* __restrict__ gv, const float* __restrict__ bev,
                       float* __restrict__ out) {
  int t = threadIdx.x;
  int zd = t & (ZDIM - 1);
  bool isVar = t >= ZDIM;
  const float* W  = isVar ? Wv : Wm;
  const float* bb = isVar ? bv : bm;
  const float* ga = isVar ? gv : gm;
  const float* be = isVar ? bev : bem;
  float a[2];
  for (int b = 0; b < 2; ++b) {
    float acc = bb[zd];
    for (int k = 0; k < HDDIM; ++k) acc += g[b * HDDIM + k] * W[(size_t)k * ZDIM + zd];
    acc += yt[b] * W[(size_t)HDDIM * ZDIM + zd];
    a[b] = acc;
  }
  float m = 0.5f * (a[0] + a[1]);
  float d0 = a[0] - m, d1 = a[1] - m;
  float var = 0.5f * (d0 * d0 + d1 * d1);        // biased, matches torch training stats
  float sc = ga[zd] * rsqrtf(var + 1e-5f);
  for (int b = 0; b < 2; ++b) {
    float r = fmaxf(sc * (a[b] - m) + be[zd], 0.f);
    if (isVar) out[2 * ZDIM + b * ZDIM + zd] = 1.0f / (1.0f + expf(-r));
    else       out[b * ZDIM + zd] = r;
  }
}

// ---------------- host side ----------------
static void run_layer(const unsigned short* actIn, const unsigned short* Wp, const float* bias,
                      unsigned short* hlin, float* agg, float* partial, float* cninv,
                      const int* erow, const int* ecol, const float* en, const float* dinv,
                      int K, int F, unsigned short* actOut, hipStream_t stream) {
  dim3 gg((MTILES + 7) / 8, F / 64);
  k_gemm_bf16<<<gg, 256, 0, stream>>>(actIn, Wp, hlin, K, F);
  int chunks = F / 4;
  size_t t1 = (size_t)BN * chunks;
  k_agg_init<<<(unsigned)((t1 + 255) / 256), 256, 0, stream>>>(hlin, dinv, agg, F);
  size_t t2 = (size_t)BATCH * NEDGE * chunks;
  k_agg_edges<<<(unsigned)((t2 + 255) / 256), 256, 0, stream>>>(hlin, erow, ecol, en, agg, F);
  k_sumsq<<<(BATCH * F * NCHUNK + 255) / 256, 256, 0, stream>>>(agg, bias, partial, F);
  k_colinv<<<(BATCH * F + 255) / 256, 256, 0, stream>>>(partial, cninv, F);
  k_normrelu<<<(unsigned)((t1 + 255) / 256), 256, 0, stream>>>(agg, bias, cninv, actOut, F);
}

extern "C" void kernel_launch(void* const* d_in, const int* in_sizes, int n_in,
                              void* d_out, int out_size, void* d_ws, size_t ws_size,
                              hipStream_t stream) {
  (void)in_sizes; (void)n_in; (void)out_size; (void)ws_size;
  const float* x  = (const float*)d_in[0];
  const int*   ei = (const int*)d_in[1];
  const float* ew = (const float*)d_in[2];
  const float* yt = (const float*)d_in[3];
  const float* W1 = (const float*)d_in[4];  const float* b1 = (const float*)d_in[5];
  const float* W2 = (const float*)d_in[6];  const float* b2 = (const float*)d_in[7];
  const float* W3 = (const float*)d_in[8];  const float* b3 = (const float*)d_in[9];
  const float* Wm = (const float*)d_in[10]; const float* bm = (const float*)d_in[11];
  const float* gm = (const float*)d_in[12]; const float* bem = (const float*)d_in[13];
  const float* Wv = (const float*)d_in[14]; const float* bv = (const float*)d_in[15];
  const float* gv = (const float*)d_in[16]; const float* bev = (const float*)d_in[17];
  const int* erow = ei;
  const int* ecol = ei + NEDGE;

  char* ws = (char*)d_ws;
  size_t off = 0;
  auto alloc = [&](size_t bytes) -> char* {
    char* p = ws + off;
    off += (bytes + 255) & ~(size_t)255;
    return p;
  };
  float* deg            = (float*)alloc((size_t)NNODE * 4);
  float* dinv           = (float*)alloc((size_t)NNODE * 4);
  float* en             = (float*)alloc((size_t)NEDGE * 4);
  unsigned short* actA  = (unsigned short*)alloc((size_t)BN * H1DIM * 2);
  unsigned short* actB  = (unsigned short*)alloc((size_t)BN * H1DIM * 2);
  unsigned short* hlin  = (unsigned short*)alloc((size_t)BN * H1DIM * 2);
  float* agg            = (float*)alloc((size_t)BN * H1DIM * 4);
  float* partial        = (float*)alloc((size_t)BATCH * NCHUNK * H1DIM * 4);
  float* cninv          = (float*)alloc((size_t)BATCH * H1DIM * 4);
  float* gpool          = (float*)alloc((size_t)BATCH * HDDIM * 4);
  unsigned short* W1p   = (unsigned short*)alloc((size_t)XDIM * H1DIM * 2);
  unsigned short* W2p   = (unsigned short*)alloc((size_t)H1DIM * H2DIM * 2);
  unsigned short* W3p   = (unsigned short*)alloc((size_t)H2DIM * HDDIM * 2);

  const int T = 256;
  // edge normalization terms
  k_deg_init<<<(NNODE + T - 1) / T, T, 0, stream>>>(deg);
  k_deg_acc<<<(NEDGE + T - 1) / T, T, 0, stream>>>(ecol, ew, deg);
  k_dinv<<<(NNODE + T - 1) / T, T, 0, stream>>>(deg, dinv);
  k_enorm<<<(NEDGE + T - 1) / T, T, 0, stream>>>(erow, ecol, ew, dinv, en);
  // input convert + weight pre-swizzle into WMMA fragment layout
  k_cvt_x<<<(unsigned)(((size_t)BN * XDIM / 4 + T - 1) / T), T, 0, stream>>>(x, actA);
  k_pack_w<<<((H1DIM / 16) * (XDIM / 32) * 32 + T - 1) / T, T, 0, stream>>>(W1, W1p, XDIM, H1DIM);
  k_pack_w<<<((H2DIM / 16) * (H1DIM / 32) * 32 + T - 1) / T, T, 0, stream>>>(W2, W2p, H1DIM, H2DIM);
  k_pack_w<<<((HDDIM / 16) * (H2DIM / 32) * 32 + T - 1) / T, T, 0, stream>>>(W3, W3p, H2DIM, HDDIM);

  run_layer(actA, W1p, b1, hlin, agg, partial, cninv, erow, ecol, en, dinv, XDIM,  H1DIM, actB, stream);
  run_layer(actB, W2p, b2, hlin, agg, partial, cninv, erow, ecol, en, dinv, H1DIM, H2DIM, actA, stream);
  run_layer(actA, W3p, b3, hlin, agg, partial, cninv, erow, ecol, en, dinv, H2DIM, HDDIM, actB, stream);

  k_pool_partial<<<(BATCH * HDDIM * NCHUNK + T - 1) / T, T, 0, stream>>>(actB, partial);
  k_pool_final<<<(BATCH * HDDIM + T - 1) / T, T, 0, stream>>>(partial, gpool);
  k_head<<<1, 256, 0, stream>>>(gpool, yt, Wm, bm, gm, bem, Wv, bv, gv, bev, (float*)d_out);
}
